// MultiHeadAttention_30949534335531
// MI455X (gfx1250) — compile-verified, matrix operand order checked by reading, not run on hardware
//
#include <hip/hip_runtime.h>
#include <hip/hip_bf16.h>

// MultiHeadAttention forward for MI455X (gfx1250), bf16 WMMA pipeline.
// B=2, S=2048, D=1024, H=16, DK=64. Workspace requirement: 64 MiB.

#define BB   2
#define SS   2048
#define DD   1024
#define HH   16
#define DKK  64
#define MM   (BB * SS)   // 4096 rows

typedef __bf16 bf16;
typedef __attribute__((ext_vector_type(16))) __bf16 v16bf;
typedef __attribute__((ext_vector_type(8)))  float   v8f;
typedef __attribute__((ext_vector_type(4)))  int     v4i;

// ---- CDNA5 feature probes (compile-safe fallbacks) -------------------------
#if defined(__AMDGCN__) && __has_builtin(__builtin_amdgcn_global_load_async_to_lds_b128)
#define HAVE_ASYNC 1
#else
#define HAVE_ASYNC 0
#endif

#if defined(__AMDGCN__) && __has_builtin(__builtin_amdgcn_update_dpp)
#define HAVE_DPP 1
#else
#define HAVE_DPP 0
#endif

#if HAVE_ASYNC
typedef __attribute__((address_space(1))) v4i* gptr_v4i;   // global int4*
typedef __attribute__((address_space(3))) v4i* lptr_v4i;   // LDS int4*
#endif

__device__ __forceinline__ v8f vzero8() {
    v8f z = {0.f, 0.f, 0.f, 0.f, 0.f, 0.f, 0.f, 0.f};
    return z;
}

// 16-byte global -> LDS copy; async (ASYNCcnt-tracked) when available.
__device__ __forceinline__ void cp16_g2l(bf16* ldst, const bf16* gsrc) {
#if HAVE_ASYNC
    bf16* g = const_cast<bf16*>(gsrc);
    __builtin_amdgcn_global_load_async_to_lds_b128(
        (gptr_v4i)(v4i*)g, (lptr_v4i)(v4i*)ldst, 0, 0);
#else
    *(uint4*)ldst = *(const uint4*)gsrc;
#endif
}

// Drain this wave's async copies, then block barrier (copies visible to all).
__device__ __forceinline__ void stage_fence() {
#if HAVE_ASYNC
#if __has_builtin(__builtin_amdgcn_s_wait_asynccnt)
    __builtin_amdgcn_s_wait_asynccnt(0);
#else
    asm volatile("s_wait_asynccnt 0x0" ::: "memory");
#endif
#endif
    __syncthreads();
}

// CDNA5 bf16 16x32 (A) / 32x16 (B) fragment loader from a row-major region.
// Layout per ISA 7.12.2: lane<16 holds K = {0..7, 16..23}, lane>=16 holds
// K = {8..15, 24..31} (kb = 8*(lane>>4)); "row" is M (A) or N (B) index.
__device__ __forceinline__ v16bf load_frag16(const bf16* p, int ld, int row) {
    const int lane = threadIdx.x & 31;
    const int kb = (lane >> 4) << 3;               // 0 or 8
    const bf16* r = p + (size_t)row * ld;
    union { v16bf v; uint4 u[2]; } f;
    f.u[0] = *(const uint4*)(r + kb);              // K = kb .. kb+7
    f.u[1] = *(const uint4*)(r + 16 + kb);         // K = 16+kb .. 16+kb+7
    return f.v;
}

__device__ __forceinline__ v8f wmma_bf16(v16bf a, v16bf b, v8f c) {
    // (neg_a, A, neg_b, B, c_mod, C, reuse_a, reuse_b)
    return __builtin_amdgcn_wmma_f32_16x16x32_bf16(false, a, false, b,
                                                   (short)0, c, false, false);
}

// ---- 16-lane (half-wave) butterfly reductions ------------------------------
// DPP version keeps reductions in the VALU pipe (dual-issue capable) instead
// of hammering the LDS pipe with ds_bpermute; rows 0-7 live in lanes 0-15 and
// rows 8-15 in lanes 16-31, and quad_perm/row_ror never cross the 16-lane row.
#if HAVE_DPP
#define DPPF(v, ctrl)                                                       \
    __builtin_bit_cast(float, __builtin_amdgcn_update_dpp(                  \
        0, __builtin_bit_cast(int, (v)), (ctrl), 0xF, 0xF, true))
__device__ __forceinline__ float half_max16(float v) {
    v = fmaxf(v, DPPF(v, 0xB1));    // quad_perm(1,0,3,2)  : xor 1
    v = fmaxf(v, DPPF(v, 0x4E));    // quad_perm(2,3,0,1)  : xor 2
    v = fmaxf(v, DPPF(v, 0x124));   // row_ror:4
    v = fmaxf(v, DPPF(v, 0x128));   // row_ror:8
    return v;
}
__device__ __forceinline__ float half_sum16(float v) {
    v += DPPF(v, 0xB1);
    v += DPPF(v, 0x4E);
    v += DPPF(v, 0x124);
    v += DPPF(v, 0x128);
    return v;
}
#else
__device__ __forceinline__ float half_max16(float v) {
    #pragma unroll
    for (int m = 1; m < 16; m <<= 1) v = fmaxf(v, __shfl_xor(v, m, 32));
    return v;
}
__device__ __forceinline__ float half_sum16(float v) {
    #pragma unroll
    for (int m = 1; m < 16; m <<= 1) v += __shfl_xor(v, m, 32);
    return v;
}
#endif

// ---------------------------------------------------------------------------
// f32 -> bf16 conversion (n must be a multiple of 4; all our sizes are)
// ---------------------------------------------------------------------------
__global__ void cvt_f32_to_bf16(const float* __restrict__ src,
                                bf16* __restrict__ dst, int n) {
    int i = (blockIdx.x * blockDim.x + threadIdx.x) * 4;
    if (i < n) {
        float4 f = *(const float4*)(src + i);
        dst[i + 0] = (bf16)f.x;
        dst[i + 1] = (bf16)f.y;
        dst[i + 2] = (bf16)f.z;
        dst[i + 3] = (bf16)f.w;
    }
}

// ---------------------------------------------------------------------------
// C[M,N] = A[M,K] * W[N,K]^T   (nn.Linear, bias=false)
// Block: 256 threads = 8 waves (4x2), block tile 128x64, wave tile 32x32.
// The block's 64-row W tile is shared by all 4 M-waves, so it is staged into
// LDS once per k-step via async global->LDS (padded stride: conflict-free).
// ---------------------------------------------------------------------------
#define WT 40   // padded LDS row stride (bf16) for the W tile

template <bool OUT_BF16>
__global__ __launch_bounds__(256)
void gemm_xwT(const bf16* __restrict__ A, const bf16* __restrict__ W,
              void* __restrict__ out, int M, int N, int K) {
    __shared__ alignas(16) bf16 Wt[64 * WT];   // [n_local=64][k=32]

    const int tid = threadIdx.x;
    const int lane = tid & 31;
    const int wave = tid >> 5;             // 0..7
    const int wm = wave & 3;               // 0..3
    const int wn = wave >> 2;              // 0..1
    const int m0 = blockIdx.y * 128 + wm * 32;
    const int nb = blockIdx.x * 64;
    const int r = lane & 15;

    // per-thread staging slice: 256 threads x 16B covers the 64x32 tile
    const int cn = tid >> 2;               // 0..63   local n row
    const int ck = (tid & 3) << 3;         // 0,8,16,24
    const bf16* wsrc = W + (size_t)(nb + cn) * K + ck;
    bf16* wdst = Wt + cn * WT + ck;

    v8f acc[2][2];
    acc[0][0] = vzero8(); acc[0][1] = vzero8();
    acc[1][0] = vzero8(); acc[1][1] = vzero8();

    for (int k0 = 0; k0 < K; k0 += 32) {
        __syncthreads();                   // previous Wt reads complete
        cp16_g2l(wdst, wsrc + k0);
        stage_fence();                     // asynccnt drain + barrier

        v16bf a0 = load_frag16(A + (size_t)m0 * K + k0, K, r);
        v16bf a1 = load_frag16(A + (size_t)(m0 + 16) * K + k0, K, r);
        v16bf b0 = load_frag16(Wt + (size_t)(wn * 32) * WT, WT, r);
        v16bf b1 = load_frag16(Wt + (size_t)(wn * 32 + 16) * WT, WT, r);
        acc[0][0] = wmma_bf16(a0, b0, acc[0][0]);
        acc[0][1] = wmma_bf16(a0, b1, acc[0][1]);
        acc[1][0] = wmma_bf16(a1, b0, acc[1][0]);
        acc[1][1] = wmma_bf16(a1, b1, acc[1][1]);
    }

    const int col = lane & 15;
    const int rb = (lane >> 4) << 3;   // lanes 0-15 -> rows 0-7, 16-31 -> 8-15
    #pragma unroll
    for (int ti = 0; ti < 2; ++ti)
        #pragma unroll
        for (int tj = 0; tj < 2; ++tj)
            #pragma unroll
            for (int i = 0; i < 8; ++i) {
                const size_t row = (size_t)(m0 + ti * 16 + rb + i);
                const int c = nb + (wn * 32) + tj * 16 + col;
                if (OUT_BF16)
                    ((bf16*)out)[row * N + c] = (bf16)acc[ti][tj][i];
                else
                    ((float*)out)[row * N + c] = acc[ti][tj][i];
            }
}

// ---------------------------------------------------------------------------
// Flash-style causal attention.
// Grid: (S/64, H, B).  Block: 128 threads = 4 waves, each wave owns 16 queries.
// Key loop in steps of 32. LDS: K tile (async-staged, shared by all 4 waves),
// V^T tile (transposed through VGPRs), per-wave P tile. All strides padded.
// ---------------------------------------------------------------------------
#define VT 40   // padded row stride (bf16) for V^T tile
#define PT 40   // padded row stride for P tile
#define KT 72   // padded row stride for K tile ([key=32][d=64])

__global__ __launch_bounds__(128)
void attn_fwd(const bf16* __restrict__ Qp, const bf16* __restrict__ Kp,
              const bf16* __restrict__ Vp, bf16* __restrict__ Ctx) {
    __shared__ alignas(16) bf16 Vt[64 * VT];       // V^T: [d=64][k=32]
    __shared__ alignas(16) bf16 Pb[4 * 16 * PT];   // per-wave P: [q=16][k=32]
    __shared__ alignas(16) bf16 Kt[32 * KT];       // K tile: [key=32][d=64]

    const int tid = threadIdx.x;
    const int lane = tid & 31;
    const int wave = tid >> 5;                     // 0..3
    const int q0 = blockIdx.x * 64;
    const int h = blockIdx.y;
    const int b = blockIdx.z;
    const int qw = q0 + wave * 16;                 // wave's first query row
    const int col = lane & 15;
    const int rb = (lane >> 4) << 3;

    // Q fragments for this wave's 16 queries (DK=64 -> two K=32 chunks)
    const bf16* qptr = Qp + ((size_t)(b * SS + qw)) * DD + h * DKK;
    const v16bf aq0 = load_frag16(qptr, DD, col);
    const v16bf aq1 = load_frag16(qptr + 32, DD, col);

    v8f o[4];
    o[0] = vzero8(); o[1] = vzero8(); o[2] = vzero8(); o[3] = vzero8();
    float mrow[8], lrow[8];
    #pragma unroll
    for (int i = 0; i < 8; ++i) { mrow[i] = -1e30f; lrow[i] = 0.f; }

    // per-thread staging slices (128 threads)
    const int kk = tid & 31;                       // key within tile
    const int db = (tid >> 5) << 4;                // 0,16,32,48 dim base

    const int kmax = q0 + 63;                      // causal upper bound (block)
    for (int kt = 0; kt <= kmax; kt += 32) {
        __syncthreads();                           // previous tile reads done

        // K tile: raw async copy, 32 keys x 64 dims (2 x 16B per thread)
        const bf16* kr = Kp + ((size_t)(b * SS + kt + kk)) * DD + h * DKK + db;
        cp16_g2l(Kt + kk * KT + db, kr);
        cp16_g2l(Kt + kk * KT + db + 8, kr + 8);

        // V^T tile: transpose must pass through VGPRs
        {
            const bf16* vr = Vp + ((size_t)(b * SS + kt + kk)) * DD + h * DKK + db;
            uint4 u0 = *(const uint4*)vr;
            uint4 u1 = *(const uint4*)(vr + 8);
            const bf16* e0 = (const bf16*)&u0;
            const bf16* e1 = (const bf16*)&u1;
            #pragma unroll
            for (int j = 0; j < 8; ++j) Vt[(db + j) * VT + kk] = e0[j];
            #pragma unroll
            for (int j = 0; j < 8; ++j) Vt[(db + 8 + j) * VT + kk] = e1[j];
        }
        stage_fence();                             // asynccnt drain + barrier

        // S = Q K^T (two 16x16 score tiles: keys kt..kt+15 and kt+16..kt+31)
        v8f s0 = vzero8(), s1 = vzero8();
        {
            v16bf b00 = load_frag16(Kt, KT, col);                 // keys 0-15, d 0-31
            v16bf b10 = load_frag16(Kt + 16 * KT, KT, col);       // keys 16-31
            s0 = wmma_bf16(aq0, b00, s0);
            s1 = wmma_bf16(aq0, b10, s1);
            v16bf b01 = load_frag16(Kt + 32, KT, col);            // d 32-63
            v16bf b11 = load_frag16(Kt + 16 * KT + 32, KT, col);
            s0 = wmma_bf16(aq1, b01, s0);
            s1 = wmma_bf16(aq1, b11, s1);
        }

        // scale 1/sqrt(DK) + causal mask (analytic; matches reference tril mask)
        #pragma unroll
        for (int i = 0; i < 8; ++i) {
            const int qi = qw + rb + i;
            float v0 = s0[i] * 0.125f;
            float v1 = s1[i] * 0.125f;
            if (kt + col > qi)      v0 = -1e30f;
            if (kt + 16 + col > qi) v1 = -1e30f;
            s0[i] = v0; s1[i] = v1;
        }

        // online softmax update + stash P in LDS (A-fragment friendly layout)
        bf16* Pw = Pb + wave * 16 * PT;
        #pragma unroll
        for (int i = 0; i < 8; ++i) {
            const float mt = half_max16(fmaxf(s0[i], s1[i]));
            const float mn = fmaxf(mrow[i], mt);
            const float al = __expf(mrow[i] - mn);
            const float p0 = __expf(s0[i] - mn);
            const float p1 = __expf(s1[i] - mn);
            const float rs = half_sum16(p0 + p1);
            lrow[i] = lrow[i] * al + rs;
            mrow[i] = mn;
            o[0][i] *= al; o[1][i] *= al; o[2][i] *= al; o[3][i] *= al;
            Pw[(rb + i) * PT + col]      = (bf16)p0;
            Pw[(rb + i) * PT + col + 16] = (bf16)p1;
        }

        // O += P(16x32) * V(32x64) : 4 WMMAs over d-tiles
        const v16bf ap = load_frag16(Pw, PT, col);
        #pragma unroll
        for (int t = 0; t < 4; ++t) {
            v16bf bv = load_frag16(Vt + (size_t)(t * 16) * VT, VT, col);
            o[t] = wmma_bf16(ap, bv, o[t]);
        }
    }

    // normalize and write context (bf16) back at this head's columns
    #pragma unroll
    for (int t = 0; t < 4; ++t)
        #pragma unroll
        for (int i = 0; i < 8; ++i) {
            const float val = o[t][i] / lrow[i];
            Ctx[((size_t)(b * SS + qw + rb + i)) * DD + h * DKK + t * 16 + col] =
                (bf16)val;
        }
}

// ---------------------------------------------------------------------------
// Host-side launch
// ---------------------------------------------------------------------------
extern "C" void kernel_launch(void* const* d_in, const int* in_sizes, int n_in,
                              void* d_out, int out_size, void* d_ws, size_t ws_size,
                              hipStream_t stream) {
    (void)in_sizes; (void)n_in; (void)out_size; (void)ws_size;

    const float* q  = (const float*)d_in[0];
    const float* k  = (const float*)d_in[1];
    const float* v  = (const float*)d_in[2];
    // d_in[3] = causal mask (tril) — applied analytically in the kernel
    const float* wq = (const float*)d_in[4];
    const float* wk = (const float*)d_in[5];
    const float* wv = (const float*)d_in[6];
    const float* wo = (const float*)d_in[7];

    const size_t NQ = (size_t)MM * DD;   // 4,194,304 activation elements
    const size_t NW = (size_t)DD * DD;   // 1,048,576 weight elements

    // Workspace layout (bf16): 7*NQ + 4*NW elements = 64 MiB total
    bf16* qb  = (bf16*)d_ws;
    bf16* kb  = qb  + NQ;
    bf16* vb  = kb  + NQ;
    bf16* Qp  = vb  + NQ;
    bf16* Kp  = Qp  + NQ;
    bf16* Vp  = Kp  + NQ;
    bf16* Ctx = Vp  + NQ;
    bf16* wqb = Ctx + NQ;
    bf16* wkb = wqb + NW;
    bf16* wvb = wkb + NW;
    bf16* wob = wvb + NW;

    const int thr = 256;
    auto cvt = [&](const float* s, bf16* d, size_t n) {
        cvt_f32_to_bf16<<<(unsigned)((n / 4 + thr - 1) / thr), thr, 0, stream>>>(
            s, d, (int)n);
    };
    cvt(q,  qb,  NQ);
    cvt(k,  kb,  NQ);
    cvt(v,  vb,  NQ);
    cvt(wq, wqb, NW);
    cvt(wk, wkb, NW);
    cvt(wv, wvb, NW);
    cvt(wo, wob, NW);

    dim3 ggrid(DD / 64, MM / 128);       // (16, 32)
    gemm_xwT<true><<<ggrid, 256, 0, stream>>>(qb, wqb, Qp, MM, DD, DD);
    gemm_xwT<true><<<ggrid, 256, 0, stream>>>(kb, wkb, Kp, MM, DD, DD);
    gemm_xwT<true><<<ggrid, 256, 0, stream>>>(vb, wvb, Vp, MM, DD, DD);

    dim3 agrid(SS / 64, HH, BB);         // (32, 16, 2)
    attn_fwd<<<agrid, 128, 0, stream>>>(Qp, Kp, Vp, Ctx);

    gemm_xwT<false><<<ggrid, 256, 0, stream>>>(Ctx, wob, d_out, MM, DD, DD);
}
